// LePE_CAT_17652315586806
// MI455X (gfx1250) — matrix operand order repaired and checked by reading, not compiled, not run on hardware
//
#include <hip/hip_runtime.h>
#include <hip/hip_bf16.h>
#include <math.h>

// ---------------------------------------------------------------------------
// LePE/CAT windowed channel attention for MI455X (gfx1250, wave32, WMMA).
// All GEMMs run on v_wmma_f32_16x16x32_bf16 (bf16 in, f32 accumulate).
// f32->bf16 conversions use native __bf16 casts (hardware v_cvt, not bit ops).
// ---------------------------------------------------------------------------

typedef __attribute__((ext_vector_type(16))) __bf16 v16bf;
typedef __attribute__((ext_vector_type(8)))  float  v8f;

struct __attribute__((aligned(16), may_alias)) U4 { unsigned v[4]; };

union Frag16 {
  v16bf v;
  U4 q4[2];
  __bf16 h[16];
};

namespace {
constexpr int Bn  = 2;
constexpr int Hh  = 192;
constexpr int Ww  = 192;
constexpr int Cc  = 96;
constexpr int NWH = 24;
constexpr int NWW = 24;
constexpr int NW  = 576;   // windows
constexpr int P2  = 64;    // pixels per window
constexpr int Ll  = Hh * Ww;   // 36864
constexpr int C3  = 3 * Cc;    // 288
constexpr int ML  = Bn * Ll;   // 73728 pixel-rows for the 1x1-conv GEMMs
}

__device__ __forceinline__ __bf16 tobf(float f) { return (__bf16)f; }

__device__ __forceinline__ v8f wmma_bf16(const Frag16& a, const Frag16& b, v8f c) {
  // D = A(16x32 bf16) * B(32x16 bf16) + C(16x16 f32)
  return __builtin_amdgcn_wmma_f32_16x16x32_bf16(
      /*neg_a=*/false, a.v, /*neg_b=*/false, b.v,
      /*c_mod=*/(short)0, c, /*reuse_a=*/false, /*reuse_b=*/false);
}

// ---------------------------------------------------------------------------
// Kernel 1: qkv 1x1 conv as GEMM  out[m, o] = sum_c x[m, c] * qkv_w[o, c]
// m indexes (b, h*W+w); out stored channels-last [ML][3C] f32.
// ---------------------------------------------------------------------------
__global__ __launch_bounds__(256) void qkv_gemm(const float* __restrict__ x,
                                                const float* __restrict__ qkv_w,
                                                float* __restrict__ qkv_lin) {
  const int lane = threadIdx.x & 31;
  const int wv   = threadIdx.x >> 5;
  const int wid  = blockIdx.x * 8 + wv;
  const int nt   = wid % (C3 / 16);
  const int mt   = wid / (C3 / 16);
  const int half = lane >> 4, lm = lane & 15;
  const int m0 = mt * 16, n0 = nt * 16;

  v8f acc;
  #pragma unroll
  for (int i = 0; i < 8; ++i) acc[i] = 0.f;

  #pragma unroll
  for (int kc = 0; kc < 3; ++kc) {
    Frag16 af, bf_;
    // A: row (m0+lm), K chunk [32*kc, 32*kc+32)
    const float* arow = x + (size_t)(m0 + lm) * Cc + kc * 32;
    #pragma unroll
    for (int i = 0; i < 8; ++i) {
      af.h[i]     = tobf(arow[16 * half + i]);       // K = 16*half + i
      af.h[8 + i] = tobf(arow[16 * half + 8 + i]);   // K = 16*half + 8 + i
    }
    // B^T row = qkv_w[n], contiguous 16 floats
    const float* brow = qkv_w + (size_t)(n0 + lm) * Cc + kc * 32 + 16 * half;
    #pragma unroll
    for (int i = 0; i < 16; ++i) bf_.h[i] = tobf(brow[i]);
    acc = wmma_bf16(af, bf_, acc);
  }
  #pragma unroll
  for (int r = 0; r < 8; ++r)
    qkv_lin[(size_t)(m0 + r + 8 * half) * C3 + n0 + lm] = acc[r];
}

// ---------------------------------------------------------------------------
// Kernel 2/4/5: depthwise 3x3 SAME, channels-last; optional exact gelu.
// blockDim.x == number of channels handled (288 or 96), grid = B*H*W.
// ---------------------------------------------------------------------------
template <bool GELU>
__global__ void dwconv3x3_k(const float* __restrict__ in, int in_stride, int in_off,
                            const float* __restrict__ w9,
                            float* __restrict__ out, int out_stride) {
  const int hwb = blockIdx.x;                // b*L + hw
  const int b = hwb / Ll, hw = hwb % Ll;
  const int h = hw / Ww, wcol = hw % Ww;
  const int o = threadIdx.x;                 // channel
  float acc = 0.f;
  #pragma unroll
  for (int dy = 0; dy < 3; ++dy) {
    const int hh = h + dy - 1;
    if ((unsigned)hh >= (unsigned)Hh) continue;
    #pragma unroll
    for (int dx = 0; dx < 3; ++dx) {
      const int ww = wcol + dx - 1;
      if ((unsigned)ww >= (unsigned)Ww) continue;
      acc += in[((size_t)b * Ll + hh * Ww + ww) * in_stride + in_off + o] *
             w9[o * 9 + dy * 3 + dx];
    }
  }
  if (GELU) acc = 0.5f * acc * (1.f + erff(acc * 0.70710678118654752f));
  out[(size_t)hwb * out_stride + o] = acc;
}

// ---------------------------------------------------------------------------
// Kernel 3: patchify + l2norm -> bf16.  qn/kn: [bc][win][64]; v transposed:
// vt: [bc][pix][win] so PV WMMA B-frags are contiguous.
// block = 64 threads (one window-channel), grid = B*C*NW.
// ---------------------------------------------------------------------------
__global__ __launch_bounds__(64) void patch_norm(const float* __restrict__ qkv_dw,
                                                 __bf16* __restrict__ qn,
                                                 __bf16* __restrict__ kn,
                                                 __bf16* __restrict__ vt) {
  const int gid = blockIdx.x;
  const int bc = gid / NW, win = gid % NW;
  const int b = bc / Cc, c = bc % Cc;
  const int p = threadIdx.x;
  const int wy = win / NWW, wx = win % NWW;
  const int py = p >> 3, px = p & 7;
  const size_t hw = (size_t)(wy * 8 + py) * Ww + (wx * 8 + px);
  const float* base = qkv_dw + ((size_t)b * Ll + hw) * C3;
  const float qv = base[c], kv = base[Cc + c], vv = base[2 * Cc + c];

  float q2 = qv * qv, k2 = kv * kv;
  #pragma unroll
  for (int m = 1; m < 32; m <<= 1) {
    q2 += __shfl_xor(q2, m, 32);
    k2 += __shfl_xor(k2, m, 32);
  }
  __shared__ float sq[2], sk[2];
  const int wvi = threadIdx.x >> 5;
  if ((threadIdx.x & 31) == 0) { sq[wvi] = q2; sk[wvi] = k2; }
  __syncthreads();
  const float qd = 1.f / fmaxf(sqrtf(sq[0] + sq[1]), 1e-12f);
  const float kd = 1.f / fmaxf(sqrtf(sk[0] + sk[1]), 1e-12f);

  const size_t ob = ((size_t)bc * NW + win) * P2 + p;
  qn[ob] = tobf(qv * qd);
  kn[ob] = tobf(kv * kd);
  vt[((size_t)bc * P2 + p) * NW + win] = tobf(vv);
}

// ---------------------------------------------------------------------------
// Kernel 6: flash-style windowed attention.  One wave per (bc, 16-window
// M-tile).  S = Qn*Kn^T (WMMA), online softmax with shfl_xor row reductions,
// P re-laid-out to an A-frag through per-wave LDS (same-wave LDS is in-order),
// O += P*V (WMMA).  Output written unpatchified, channels-last into xo.
// ---------------------------------------------------------------------------
__global__ __launch_bounds__(256) void attn_win(const __bf16* __restrict__ qn,
                                                const __bf16* __restrict__ kn,
                                                const __bf16* __restrict__ vtr,
                                                const float* __restrict__ scale_p,
                                                float* __restrict__ xo) {
  __shared__ __align__(16) __bf16 ldsP[8][16 * 32];  // per-wave P tile
  const int lane = threadIdx.x & 31;
  const int wv   = threadIdx.x >> 5;
  const int wid  = blockIdx.x * 8 + wv;
  const int bc = wid / (NW / 16);
  const int mt = wid % (NW / 16);
  const int half = lane >> 4, lm = lane & 15;
  const float scale = scale_p[0];

  const __bf16* q = qn  + (size_t)bc * NW * P2;
  const __bf16* k = kn  + (size_t)bc * NW * P2;
  const __bf16* v = vtr + (size_t)bc * P2 * NW;

  Frag16 qa[2];
  {
    const __bf16* qrow = q + (size_t)(mt * 16 + lm) * P2;
    #pragma unroll
    for (int f = 0; f < 2; ++f) {
      qa[f].q4[0] = *(const U4*)(qrow + 32 * f + 16 * half);
      qa[f].q4[1] = *(const U4*)(qrow + 32 * f + 16 * half + 8);
    }
  }

  v8f o[4];
  #pragma unroll
  for (int t = 0; t < 4; ++t)
    #pragma unroll
    for (int r = 0; r < 8; ++r) o[t][r] = 0.f;
  float mrow[8], lrow[8];
  #pragma unroll
  for (int r = 0; r < 8; ++r) { mrow[r] = -3.0e38f; lrow[r] = 0.f; }

  __bf16* myP = &ldsP[wv][0];

  for (int kt = 0; kt < NW / 32; ++kt) {
    const int kb = kt * 32;
    v8f s[2];
    #pragma unroll
    for (int j = 0; j < 2; ++j) {
      const __bf16* krow = k + (size_t)(kb + 16 * j + lm) * P2;
      Frag16 b0, b1;
      b0.q4[0] = *(const U4*)(krow + 16 * half);
      b0.q4[1] = *(const U4*)(krow + 16 * half + 8);
      b1.q4[0] = *(const U4*)(krow + 32 + 16 * half);
      b1.q4[1] = *(const U4*)(krow + 32 + 16 * half + 8);
      v8f z;
      #pragma unroll
      for (int i = 0; i < 8; ++i) z[i] = 0.f;
      z = wmma_bf16(qa[0], b0, z);
      z = wmma_bf16(qa[1], b1, z);
      #pragma unroll
      for (int r = 0; r < 8; ++r) s[j][r] = z[r] * scale;
    }

    // online softmax per output row (r + 8*half); columns live across 16 lanes
    #pragma unroll
    for (int r = 0; r < 8; ++r) {
      float rm = fmaxf(s[0][r], s[1][r]);
      rm = fmaxf(rm, __shfl_xor(rm, 1, 32));
      rm = fmaxf(rm, __shfl_xor(rm, 2, 32));
      rm = fmaxf(rm, __shfl_xor(rm, 4, 32));
      rm = fmaxf(rm, __shfl_xor(rm, 8, 32));
      const float mn   = fmaxf(mrow[r], rm);
      const float corr = __expf(mrow[r] - mn);
      const float p0 = __expf(s[0][r] - mn);
      const float p1 = __expf(s[1][r] - mn);
      float ps = p0 + p1;
      ps += __shfl_xor(ps, 1, 32);
      ps += __shfl_xor(ps, 2, 32);
      ps += __shfl_xor(ps, 4, 32);
      ps += __shfl_xor(ps, 8, 32);
      lrow[r] = lrow[r] * corr + ps;
      mrow[r] = mn;
      o[0][r] *= corr; o[1][r] *= corr; o[2][r] *= corr; o[3][r] *= corr;
      myP[(r + 8 * half) * 32 + lm]      = tobf(p0);
      myP[(r + 8 * half) * 32 + 16 + lm] = tobf(p1);
    }
    asm volatile("" ::: "memory");   // keep LDS stores ahead of the re-loads
    Frag16 pa;                       // A-frag of P: row lm, K chunks per half
    pa.q4[0] = *(const U4*)(myP + lm * 32 + 16 * half);
    pa.q4[1] = *(const U4*)(myP + lm * 32 + 16 * half + 8);
    asm volatile("" ::: "memory");

    #pragma unroll
    for (int t = 0; t < 4; ++t) {
      const __bf16* vrow = v + (size_t)(16 * t + lm) * NW + kb + 16 * half;
      Frag16 vb;
      vb.q4[0] = *(const U4*)(vrow);
      vb.q4[1] = *(const U4*)(vrow + 8);
      o[t] = wmma_bf16(pa, vb, o[t]);
    }
  }

  const int b = bc / Cc, c = bc % Cc;
  float inv[8];
  #pragma unroll
  for (int r = 0; r < 8; ++r) inv[r] = 1.f / lrow[r];
  #pragma unroll
  for (int t = 0; t < 4; ++t) {
    const int pixel = 16 * t + lm;
    const int py = pixel >> 3, px = pixel & 7;
    #pragma unroll
    for (int r = 0; r < 8; ++r) {
      const int win = mt * 16 + r + 8 * half;
      const int wy = win / NWW, wx = win % NWW;
      const float val = o[t][r] * inv[r];
      xo[(((size_t)b * Ll) + (size_t)(wy * 8 + py) * Ww + (wx * 8 + px)) * Cc + c] = val;
    }
  }
}

// ---------------------------------------------------------------------------
// Kernel 7: proj 1x1 conv (WMMA) + bias + pe, stored to d_out [B, L, C] f32.
// C accumulator preloaded with pe + bias.
// ---------------------------------------------------------------------------
__global__ __launch_bounds__(256) void proj_out(const float* __restrict__ xo,
                                                const float* __restrict__ proj_w,
                                                const float* __restrict__ proj_b,
                                                const float* __restrict__ pe,
                                                float* __restrict__ out) {
  const int lane = threadIdx.x & 31;
  const int wv   = threadIdx.x >> 5;
  const int wid  = blockIdx.x * 8 + wv;
  const int nt = wid % (Cc / 16);
  const int mt = wid / (Cc / 16);
  const int half = lane >> 4, lm = lane & 15;
  const int m0 = mt * 16, n = nt * 16 + lm;

  v8f acc;
  const float bias = proj_b[n];
  #pragma unroll
  for (int r = 0; r < 8; ++r)
    acc[r] = pe[(size_t)(m0 + r + 8 * half) * Cc + n] + bias;

  #pragma unroll
  for (int kc = 0; kc < 3; ++kc) {
    Frag16 af, bf_;
    const float* arow = xo + (size_t)(m0 + lm) * Cc + kc * 32;
    #pragma unroll
    for (int i = 0; i < 8; ++i) {
      af.h[i]     = tobf(arow[16 * half + i]);
      af.h[8 + i] = tobf(arow[16 * half + 8 + i]);
    }
    const float* brow = proj_w + (size_t)n * Cc + kc * 32 + 16 * half;
    #pragma unroll
    for (int i = 0; i < 16; ++i) bf_.h[i] = tobf(brow[i]);
    acc = wmma_bf16(af, bf_, acc);
  }
  #pragma unroll
  for (int r = 0; r < 8; ++r)
    out[(size_t)(m0 + r + 8 * half) * Cc + n] = acc[r];
}

// ---------------------------------------------------------------------------
extern "C" void kernel_launch(void* const* d_in, const int* in_sizes, int n_in,
                              void* d_out, int out_size, void* d_ws, size_t ws_size,
                              hipStream_t stream) {
  (void)in_sizes; (void)n_in; (void)out_size; (void)ws_size;
  const float* x        = (const float*)d_in[0];
  const float* qkv_w    = (const float*)d_in[1];
  const float* qkv_dw_w = (const float*)d_in[2];
  const float* scale    = (const float*)d_in[3];
  const float* proj_w   = (const float*)d_in[4];
  const float* proj_b   = (const float*)d_in[5];
  const float* pe_w1    = (const float*)d_in[6];
  const float* pe_w2    = (const float*)d_in[7];
  float* out = (float*)d_out;

  char* ws = (char*)d_ws;
  size_t off = 0;
  auto alloc = [&](size_t bytes) -> char* {
    char* p = ws + off;
    off += (bytes + 255) & ~(size_t)255;
    return p;
  };
  float*  qkv_lin = (float*)alloc((size_t)ML * C3 * sizeof(float));
  float*  qkv_dw  = (float*)alloc((size_t)ML * C3 * sizeof(float));
  __bf16* qn = (__bf16*)alloc((size_t)Bn * Cc * NW * P2 * 2);
  __bf16* kn = (__bf16*)alloc((size_t)Bn * Cc * NW * P2 * 2);
  __bf16* vt = (__bf16*)alloc((size_t)Bn * Cc * NW * P2 * 2);
  float*  xo = (float*)alloc((size_t)ML * Cc * sizeof(float));
  // pe1 / pe reuse the qkv_lin region (fully consumed by the dwconv kernel
  // before the pe kernels run; stream order guarantees safety).
  float* pe1 = qkv_lin;
  float* pe  = qkv_lin + (size_t)ML * Cc;

  // 1) qkv = x @ qkv_w^T                   (WMMA bf16)
  qkv_gemm<<<(ML / 16) * (C3 / 16) / 8, 256, 0, stream>>>(x, qkv_w, qkv_lin);
  // 2) depthwise 3x3 over all 3C channels
  dwconv3x3_k<false><<<ML, C3, 0, stream>>>(qkv_lin, C3, 0, qkv_dw_w, qkv_dw, C3);
  // 3) patchify + l2norm + bf16 convert (q, k) and transpose (v)
  patch_norm<<<Bn * Cc * NW, 64, 0, stream>>>(qkv_dw, qn, kn, vt);
  // 4) pe1 = gelu(dwconv(v, pe_w1))
  dwconv3x3_k<true ><<<ML, Cc, 0, stream>>>(qkv_dw, C3, 2 * Cc, pe_w1, pe1, Cc);
  // 5) pe = dwconv(pe1, pe_w2)
  dwconv3x3_k<false><<<ML, Cc, 0, stream>>>(pe1, Cc, 0, pe_w2, pe, Cc);
  // 6) windowed attention (flash-style, WMMA bf16)
  attn_win<<<(Bn * Cc) * (NW / 16) / 8, 256, 0, stream>>>(qn, kn, vt, scale, xo);
  // 7) out = xo @ proj_w^T + proj_b + pe   (WMMA bf16)
  proj_out<<<(ML / 16) * (Cc / 16) / 8, 256, 0, stream>>>(xo, proj_w, proj_b, pe, out);
}
